// SelfAttentionMechanism_88098369175728
// MI455X (gfx1250) — compile-verified, matrix operand order checked by reading, not run on hardware
//
#include <hip/hip_runtime.h>
#include <math.h>

// ---------- constants ----------
#define T_SEQ 2048
#define C_EMB 1024
#define HEADS 16
#define HDIM  64
#define BATCH 2
#define ROWS  (BATCH * T_SEQ)   // 4096 token rows
#define NK    (C_EMB / 32)      // 32 K-steps for the dense GEMMs

// ---------- WMMA fragment types (gfx1250, wave32) ----------
typedef __attribute__((ext_vector_type(16))) __bf16 v16bf;
typedef __attribute__((ext_vector_type(8)))  float  v8f;

union FragAB { v16bf v; uint4 q[2]; unsigned int d[8]; unsigned short u[16]; };
union FragC  { v8f v; float f[8]; };

__device__ __forceinline__ unsigned short f2bf(float x) {
  unsigned int u = __float_as_uint(x);
  u += 0x7fffu + ((u >> 16) & 1u);           // round-to-nearest-even
  return (unsigned short)(u >> 16);
}

// 16x32 bf16 fragment from a row-major (or K-major transposed) array.
// ISA layout (CDNA5 ISA): lanes 0-15 row lm, K {0..7,16..23}; lanes 16-31 row lm,
// K {8..15,24..31}.  Two 16B loads per lane.  Requires ld % 8 == 0.
__device__ __forceinline__ FragAB load_ab(const unsigned short* tile, int ld) {
  int lane = threadIdx.x & 31;
  int hf   = lane >> 4;
  int lm   = lane & 15;
  const unsigned short* p = tile + (size_t)lm * ld + 8 * hf;
  FragAB f;
  f.q[0] = *reinterpret_cast<const uint4*>(p);        // K 0..7   (+8*hf)
  f.q[1] = *reinterpret_cast<const uint4*>(p + 16);   // K 16..23 (+8*hf)
  return f;
}

// Same fragment but from a 16x32 bf16 tile staged in LDS (row stride 32 elems).
__device__ __forceinline__ FragAB load_a_lds(const unsigned short* lbase) {
  int lane = threadIdx.x & 31;
  int hf   = lane >> 4;
  int lm   = lane & 15;
  const unsigned short* p = lbase + lm * 32 + 8 * hf;
  FragAB f;
  f.q[0] = *reinterpret_cast<const uint4*>(p);
  f.q[1] = *reinterpret_cast<const uint4*>(p + 16);
  return f;
}

__device__ __forceinline__ v8f wmma_bf16(const FragAB& a, const FragAB& b, v8f c) {
  return __builtin_amdgcn_wmma_f32_16x16x32_bf16(false, a.v, false, b.v,
                                                 (short)0, c, false, false);
}

// Async-stage a 16x32 bf16 tile (1KB) global -> LDS, 16B per lane, 2 instructions.
// LDS layout: row-major, 64B row stride.  Issued by one wave; tracked on ASYNCcnt.
__device__ __forceinline__ void async_stage_a(const unsigned short* gbase, int ld,
                                              unsigned int ldsbase /*byte addr*/) {
  int lane = threadIdx.x & 31;
  int r0 = lane >> 2;                 // rows 0..7
  int c0 = (lane & 3) * 8;            // 4 x 16B chunks per 64B row
  unsigned long long g0 = (unsigned long long)(gbase + (size_t)r0 * ld + c0);
  unsigned long long g1 = (unsigned long long)(gbase + (size_t)(r0 + 8) * ld + c0);
  unsigned int l0 = ldsbase + (unsigned int)lane * 16u;
  unsigned int l1 = l0 + 512u;
  asm volatile("global_load_async_to_lds_b128 %0, %1, off" :: "v"(l0), "v"(g0) : "memory");
  asm volatile("global_load_async_to_lds_b128 %0, %1, off" :: "v"(l1), "v"(g1) : "memory");
}

// ---------- conversion kernels ----------
__global__ void k_f32_to_bf16(const float* __restrict__ src,
                              unsigned short* __restrict__ dst, int n) {
  int i = blockIdx.x * blockDim.x + threadIdx.x;
  if (i < n) dst[i] = f2bf(src[i]);
}

// src[rows][cols] (f32) -> dst[cols][rows] (bf16)
__global__ void k_transpose_to_bf16(const float* __restrict__ src,
                                    unsigned short* __restrict__ dst,
                                    int rows, int cols) {
  int i = blockIdx.x * blockDim.x + threadIdx.x;
  if (i < rows * cols) {
    int r = i / cols, c = i % cols;
    dst[(size_t)c * rows + r] = f2bf(src[i]);
  }
}

// ---------- QKV GEMM: block = 16 rows x 512 cols, wave = 16x64 ----------
// A strip (16 x 32 per K-step) async-staged to LDS once per block, double-buffered.
__global__ void k_gemm_qkv(const unsigned short* __restrict__ xb,     // [4096][1024]
                           const unsigned short* __restrict__ WattT,  // [3072][1024]
                           const float* __restrict__ b_attn,
                           unsigned short* __restrict__ Qb,   // [B][H][T][D]
                           unsigned short* __restrict__ Kb,   // [B][H][T][D]
                           unsigned short* __restrict__ Vtb)  // [B][H][D][T]
{
  __shared__ __align__(16) unsigned short abuf[2][16][32];   // 2 x 1KB
  const int colBlocks = (3 * C_EMB) / 512;   // 6
  int mt = blockIdx.x / colBlocks;
  int cb = blockIdx.x % colBlocks;
  int wib = threadIdx.x >> 5;
  int lane = threadIdx.x & 31, hf = lane >> 4, lm = lane & 15;
  int colbase = cb * 512 + wib * 64;

  const unsigned short* Abase = xb + (size_t)(mt * 16) * C_EMB;
  const unsigned short* Bb[4];
#pragma unroll
  for (int nt = 0; nt < 4; ++nt)
    Bb[nt] = WattT + (size_t)(colbase + nt * 16) * C_EMB;

  unsigned int lds0 = (unsigned int)(size_t)&abuf[0][0][0];
  bool stager = (wib == 0);
  if (stager) async_stage_a(Abase, C_EMB, lds0);

  FragC acc[4] = {};
  for (int kb = 0; kb < NK; ++kb) {
    if (stager) asm volatile("s_wait_asynccnt 0" ::: "memory");
    __syncthreads();                               // buf[kb&1] ready, prev reads done
    if (stager && (kb + 1) < NK)
      async_stage_a(Abase + (kb + 1) * 32, C_EMB, lds0 + ((kb + 1) & 1) * 1024u);

    FragAB a = load_a_lds(&abuf[kb & 1][0][0]);
#pragma unroll
    for (int nt = 0; nt < 4; ++nt) {
      const unsigned short* bp = Bb[nt] + kb * 32;
      __builtin_prefetch(bp + 32, 0, 1);           // global_prefetch_b8 next K-step
      FragAB b = load_ab(bp, C_EMB);
      acc[nt].v = wmma_bf16(a, b, acc[nt].v);
    }
  }

#pragma unroll
  for (int nt = 0; nt < 4; ++nt) {
    int c0 = colbase + nt * 16 + lm;       // same q/k/v section & head per tile
    float bias = b_attn[c0];
    int which = c0 / C_EMB;
    int cc = c0 % C_EMB;
    int h = cc / HDIM, d = cc % HDIM;
#pragma unroll
    for (int r = 0; r < 8; ++r) {
      int gr = mt * 16 + r + 8 * hf;       // token row in [0,4096)
      int bb = gr / T_SEQ, t = gr % T_SEQ;
      unsigned short bv = f2bf(acc[nt].f[r] + bias);
      size_t rowmaj = ((size_t)(bb * HEADS + h) * T_SEQ + t) * HDIM + d;
      if (which == 0)      Qb[rowmaj] = bv;
      else if (which == 1) Kb[rowmaj] = bv;
      else Vtb[((size_t)(bb * HEADS + h) * HDIM + d) * T_SEQ + t] = bv;
    }
  }
}

// ---------- flash attention: one wave per (b, h, 16-query block) ----------
__global__ void k_attention(const unsigned short* __restrict__ Qb,
                            const unsigned short* __restrict__ Kb,
                            const unsigned short* __restrict__ Vtb,
                            unsigned short* __restrict__ Yb)  // [B][T][C] bf16
{
  __shared__ __align__(16) float pbuf[8][16][32];   // per-wave P staging (2KB each)
  int wib = threadIdx.x >> 5;
  int wid = blockIdx.x * (blockDim.x >> 5) + wib;
  const int qblocks = T_SEQ / 16;            // 128
  int b   = wid / (HEADS * qblocks);
  int rem = wid % (HEADS * qblocks);
  int h   = rem / qblocks;
  int qb  = rem % qblocks;
  if (b >= BATCH) return;
  int q0 = qb * 16;

  int lane = threadIdx.x & 31, hf = lane >> 4, lm = lane & 15;
  const unsigned short* Qh = Qb  + (size_t)(b * HEADS + h) * T_SEQ * HDIM;
  const unsigned short* Kh = Kb  + (size_t)(b * HEADS + h) * T_SEQ * HDIM;
  const unsigned short* Vh = Vtb + (size_t)(b * HEADS + h) * HDIM * T_SEQ;

  FragAB qf0 = load_ab(Qh + (size_t)q0 * HDIM + 0,  HDIM);
  FragAB qf1 = load_ab(Qh + (size_t)q0 * HDIM + 32, HDIM);

  FragC acc[4] = {};                         // output d-tiles, 16x64 total
  float m_i[8], l_i[8];
#pragma unroll
  for (int r = 0; r < 8; ++r) { m_i[r] = -__builtin_inff(); l_i[r] = 0.0f; }

  const float scale = 0.125f;                // 1/sqrt(64)
  int kend = q0 + 16;                        // exclusive causal key bound
  for (int k0 = 0; k0 < kend; k0 += 32) {
    FragC s0 = {}, s1 = {};
    FragAB kf = load_ab(Kh + (size_t)k0 * HDIM + 0, HDIM);
    s0.v = wmma_bf16(qf0, kf, s0.v);
    kf = load_ab(Kh + (size_t)k0 * HDIM + 32, HDIM);
    s0.v = wmma_bf16(qf1, kf, s0.v);
    bool haveS1 = (k0 + 16) < kend;          // wave-uniform
    if (haveS1) {
      kf = load_ab(Kh + (size_t)(k0 + 16) * HDIM + 0, HDIM);
      s1.v = wmma_bf16(qf0, kf, s1.v);
      kf = load_ab(Kh + (size_t)(k0 + 16) * HDIM + 32, HDIM);
      s1.v = wmma_bf16(qf1, kf, s1.v);
    }

    int j0 = k0 + lm, j1 = k0 + 16 + lm;
#pragma unroll
    for (int r = 0; r < 8; ++r) {
      int t = q0 + r + 8 * hf;
      float v0 = (j0 <= t) ? s0.f[r] * scale : -__builtin_inff();
      float v1 = (haveS1 && j1 <= t) ? s1.f[r] * scale : -__builtin_inff();
      float rmax = fmaxf(v0, v1);
#pragma unroll
      for (int off = 1; off < 16; off <<= 1)
        rmax = fmaxf(rmax, __shfl_xor(rmax, off, 32));
      float mnew  = fmaxf(m_i[r], rmax);
      float alpha = __expf(m_i[r] - mnew);   // 0 on first block
      float p0 = __expf(v0 - mnew);
      float p1 = __expf(v1 - mnew);
      float rsum = p0 + p1;
#pragma unroll
      for (int off = 1; off < 16; off <<= 1)
        rsum += __shfl_xor(rsum, off, 32);
      l_i[r] = l_i[r] * alpha + rsum;
      m_i[r] = mnew;
      acc[0].f[r] *= alpha; acc[1].f[r] *= alpha;
      acc[2].f[r] *= alpha; acc[3].f[r] *= alpha;
      pbuf[wib][r + 8 * hf][lm]      = p0;   // C-layout -> LDS
      pbuf[wib][r + 8 * hf][16 + lm] = p1;
    }

    asm volatile("s_wait_dscnt 0" ::: "memory");  // P stores visible to re-read

    // rebuild P as a 16x32 bf16 A-fragment (transpose via LDS)
    FragAB pa;
    const float* pr = &pbuf[wib][lm][8 * hf];
#pragma unroll
    for (int i = 0; i < 8; ++i) {
      int kk = (i < 4) ? (2 * i) : (2 * i + 8);
      pa.u[2 * i]     = f2bf(pr[kk]);
      pa.u[2 * i + 1] = f2bf(pr[kk + 1]);
    }

#pragma unroll
    for (int nt = 0; nt < 4; ++nt) {
      FragAB vf = load_ab(Vh + (size_t)(nt * 16) * T_SEQ + k0, T_SEQ);
      acc[nt].v = wmma_bf16(pa, vf, acc[nt].v);
    }
  }

  // epilogue: normalize and store to Y[b][t][h*64 + d] (bf16)
#pragma unroll
  for (int r = 0; r < 8; ++r) {
    float rinv = 1.0f / l_i[r];
    int t = q0 + r + 8 * hf;
    size_t rowbase = ((size_t)b * T_SEQ + t) * C_EMB + h * HDIM;
#pragma unroll
    for (int nt = 0; nt < 4; ++nt)
      Yb[rowbase + nt * 16 + lm] = f2bf(acc[nt].f[r] * rinv);
  }
}

// ---------- output projection: block = 16 x 512, wave = 16x64 ----------
__global__ void k_gemm_proj(const unsigned short* __restrict__ Yb,      // [4096][1024]
                            const unsigned short* __restrict__ WprojT,  // [1024][1024]
                            const float* __restrict__ b_proj,
                            float* __restrict__ out)                    // [4096][1024]
{
  __shared__ __align__(16) unsigned short abuf[2][16][32];
  const int colBlocks = C_EMB / 512;   // 2
  int mt = blockIdx.x / colBlocks;
  int cb = blockIdx.x % colBlocks;
  int wib = threadIdx.x >> 5;
  int lane = threadIdx.x & 31, hf = lane >> 4, lm = lane & 15;
  int colbase = cb * 512 + wib * 64;

  const unsigned short* Abase = Yb + (size_t)(mt * 16) * C_EMB;
  const unsigned short* Bb[4];
#pragma unroll
  for (int nt = 0; nt < 4; ++nt)
    Bb[nt] = WprojT + (size_t)(colbase + nt * 16) * C_EMB;

  unsigned int lds0 = (unsigned int)(size_t)&abuf[0][0][0];
  bool stager = (wib == 0);
  if (stager) async_stage_a(Abase, C_EMB, lds0);

  FragC acc[4] = {};
  for (int kb = 0; kb < NK; ++kb) {
    if (stager) asm volatile("s_wait_asynccnt 0" ::: "memory");
    __syncthreads();
    if (stager && (kb + 1) < NK)
      async_stage_a(Abase + (kb + 1) * 32, C_EMB, lds0 + ((kb + 1) & 1) * 1024u);

    FragAB a = load_a_lds(&abuf[kb & 1][0][0]);
#pragma unroll
    for (int nt = 0; nt < 4; ++nt) {
      const unsigned short* bp = Bb[nt] + kb * 32;
      __builtin_prefetch(bp + 32, 0, 1);
      FragAB b = load_ab(bp, C_EMB);
      acc[nt].v = wmma_bf16(a, b, acc[nt].v);
    }
  }

#pragma unroll
  for (int nt = 0; nt < 4; ++nt) {
    float bias = b_proj[colbase + nt * 16 + lm];
#pragma unroll
    for (int r = 0; r < 8; ++r) {
      int gr = mt * 16 + r + 8 * hf;
      out[(size_t)gr * C_EMB + colbase + nt * 16 + lm] = acc[nt].f[r] + bias;
    }
  }
}

// ---------- host launcher ----------
extern "C" void kernel_launch(void* const* d_in, const int* in_sizes, int n_in,
                              void* d_out, int out_size, void* d_ws, size_t ws_size,
                              hipStream_t stream) {
  const float* x      = (const float*)d_in[0];
  const float* W_attn = (const float*)d_in[1];
  const float* b_attn = (const float*)d_in[2];
  const float* W_proj = (const float*)d_in[3];
  const float* b_proj = (const float*)d_in[4];
  float* out = (float*)d_out;

  char* ws = (char*)d_ws;
  size_t off = 0;
  auto carve = [&](size_t bytes) -> char* {
    char* p = ws + off;
    off = (off + bytes + 255) & ~(size_t)255;
    return p;
  };
  unsigned short* xb     = (unsigned short*)carve((size_t)ROWS * C_EMB * 2);
  unsigned short* WattT  = (unsigned short*)carve((size_t)3 * C_EMB * C_EMB * 2);
  unsigned short* WprojT = (unsigned short*)carve((size_t)C_EMB * C_EMB * 2);
  unsigned short* Qb     = (unsigned short*)carve((size_t)ROWS * C_EMB * 2 + 8192);
  unsigned short* Kb     = (unsigned short*)carve((size_t)ROWS * C_EMB * 2 + 8192);
  unsigned short* Vtb    = (unsigned short*)carve((size_t)ROWS * C_EMB * 2 + 8192);
  unsigned short* Yb     = (unsigned short*)carve((size_t)ROWS * C_EMB * 2);

  int n1 = ROWS * C_EMB;
  k_f32_to_bf16<<<(n1 + 255) / 256, 256, 0, stream>>>(x, xb, n1);
  int n2 = C_EMB * 3 * C_EMB;
  k_transpose_to_bf16<<<(n2 + 255) / 256, 256, 0, stream>>>(W_attn, WattT, C_EMB, 3 * C_EMB);
  int n3 = C_EMB * C_EMB;
  k_transpose_to_bf16<<<(n3 + 255) / 256, 256, 0, stream>>>(W_proj, WprojT, C_EMB, C_EMB);

  // QKV: 256 row tiles x 6 col super-blocks (16x512 per block, 8 waves)
  k_gemm_qkv<<<256 * 6, 256, 0, stream>>>(xb, WattT, b_attn, Qb, Kb, Vtb);
  // attention: B*H*(T/16) = 4096 wave-tasks, 8 waves/block
  k_attention<<<(BATCH * HEADS * (T_SEQ / 16)) / 8, 256, 0, stream>>>(Qb, Kb, Vtb, Yb);
  // proj: 256 row tiles x 2 col super-blocks
  k_gemm_proj<<<256 * 2, 256, 0, stream>>>(Yb, WprojT, b_proj, out);
}
